// SetAbstractionMSG_46299747450895
// MI455X (gfx1250) — matrix operand order repaired and checked by reading, not compiled, not run on hardware
//
#include <hip/hip_runtime.h>

typedef __attribute__((ext_vector_type(16))) _Float16 v16h;
typedef __attribute__((ext_vector_type(8)))  float    v8f;

#define B_   8
#define N_   4096
#define S_   1024
#define KP1  96      // cin=67 padded to 96 (multiple of 32)
#define MID  128
#define OUTC 256
#define CF   512
#define LDS_STRIDE 136                 // 128 + 8 halfs pad; row stride 272B (16B aligned)
#define SMEM_HALFS ((32 + 128) * LDS_STRIDE)
#define SMEM_BYTES (SMEM_HALFS * 2)

struct __align__(16) U128 { unsigned int x, y, z, w; };
union Frag { v16h v; U128 q[2]; };

// ---------------------------------------------------------------- FPS
__device__ __forceinline__ unsigned long long shflxor64(unsigned long long v, int m) {
  unsigned lo = (unsigned)(v & 0xFFFFFFFFull);
  unsigned hi = (unsigned)(v >> 32);
  lo = __shfl_xor(lo, m, 32);
  hi = __shfl_xor(hi, m, 32);
  return ((unsigned long long)hi << 32) | lo;
}

__global__ void __launch_bounds__(256) fps_kernel(const float* __restrict__ xyz,
                                                  float* __restrict__ newxyz) {
  __shared__ float px[N_], py[N_], pz[N_];
  __shared__ unsigned long long red[8];
  __shared__ int lastSh;
  const int b = blockIdx.x, tid = threadIdx.x;
  for (int j = tid; j < N_; j += 256) {
    px[j] = xyz[(size_t)(b * N_ + j) * 3 + 0];
    py[j] = xyz[(size_t)(b * N_ + j) * 3 + 1];
    pz[j] = xyz[(size_t)(b * N_ + j) * 3 + 2];
  }
  __syncthreads();
  float d[16];
#pragma unroll
  for (int t = 0; t < 16; ++t) d[t] = 1e10f;
  int last = 0;
  for (int i = 0; i < S_; ++i) {
    const float lx = px[last], ly = py[last], lz = pz[last];
    if (tid == 0) {
      newxyz[(size_t)(b * S_ + i) * 3 + 0] = lx;
      newxyz[(size_t)(b * S_ + i) * 3 + 1] = ly;
      newxyz[(size_t)(b * S_ + i) * 3 + 2] = lz;
    }
    unsigned long long best = 0ull;
#pragma unroll
    for (int t = 0; t < 16; ++t) {
      const int j = t * 256 + tid;
      const float dx = px[j] - lx, dy = py[j] - ly, dz = pz[j] - lz;
      const float dd = dx * dx + dy * dy + dz * dz;
      d[t] = fminf(d[t], dd);
      unsigned long long key =
          ((unsigned long long)__float_as_uint(d[t]) << 32) |
          (unsigned)(0xFFFFFFFFu - (unsigned)j);          // max dist, ties -> min idx
      best = best > key ? best : key;
    }
#pragma unroll
    for (int off = 16; off >= 1; off >>= 1) {
      unsigned long long o = shflxor64(best, off);
      best = best > o ? best : o;
    }
    if ((tid & 31) == 0) red[tid >> 5] = best;
    __syncthreads();
    if (tid == 0) {
      unsigned long long m = red[0];
      for (int w = 1; w < 8; ++w) m = m > red[w] ? m : red[w];
      lastSh = (int)(0xFFFFFFFFu - (unsigned)(m & 0xFFFFFFFFull));
    }
    __syncthreads();
    last = lastSh;
  }
}

// ---------------------------------------------------------------- ball query
__global__ void __launch_bounds__(256) ball_query_kernel(const float* __restrict__ xyz,
                                                         const float* __restrict__ newxyz,
                                                         int* __restrict__ idx,
                                                         float r2, int k) {
  __shared__ float tx[512], ty[512], tz[512];
  const int b = blockIdx.x;
  const int s = blockIdx.y * 256 + threadIdx.x;
  const float nx = newxyz[(size_t)(b * S_ + s) * 3 + 0];
  const float ny = newxyz[(size_t)(b * S_ + s) * 3 + 1];
  const float nz = newxyz[(size_t)(b * S_ + s) * 3 + 2];
  const size_t base = (size_t)(b * S_ + s) * k;
  int cnt = 0, first = 0;
  for (int j0 = 0; j0 < N_; j0 += 512) {
    for (int t = threadIdx.x; t < 512; t += 256) {
      tx[t] = xyz[(size_t)(b * N_ + j0 + t) * 3 + 0];
      ty[t] = xyz[(size_t)(b * N_ + j0 + t) * 3 + 1];
      tz[t] = xyz[(size_t)(b * N_ + j0 + t) * 3 + 2];
    }
    __syncthreads();
    for (int t = 0; t < 512; ++t) {
      if (cnt < k) {
        const float dx = nx - tx[t], dy = ny - ty[t], dz = nz - tz[t];
        if (dx * dx + dy * dy + dz * dz < r2) {
          if (cnt == 0) first = j0 + t;
          idx[base + cnt] = j0 + t;
          ++cnt;
        }
      }
    }
    __syncthreads();
  }
  for (int c = cnt; c < k; ++c) idx[base + c] = first;
}

// ---------------------------------------------------------------- grouped input (f16, padded to 96 ch)
__global__ void __launch_bounds__(256) build_x_kernel(const float* __restrict__ xyz,
                                                      const float* __restrict__ feat,
                                                      const float* __restrict__ newxyz,
                                                      const int* __restrict__ idx,
                                                      _Float16* __restrict__ Xh, int k) {
  const int r = blockIdx.x * 256 + threadIdx.x;
  const int M = B_ * S_ * k;
  if (r >= M) return;
  const int b = r / (S_ * k);
  const int rem = r % (S_ * k);
  const int s = rem / k;
  const int j = idx[r];
  const float* p = &xyz[(size_t)(b * N_ + j) * 3];
  const float* q = &newxyz[(size_t)(b * S_ + s) * 3];
  _Float16* o = Xh + (size_t)r * KP1;
  o[0] = (_Float16)(p[0] - q[0]);
  o[1] = (_Float16)(p[1] - q[1]);
  o[2] = (_Float16)(p[2] - q[2]);
  const float* f = &feat[((size_t)b * N_ + j) * 64];
#pragma unroll
  for (int c = 0; c < 64; ++c) o[3 + c] = (_Float16)f[c];
#pragma unroll
  for (int c = 67; c < KP1; ++c) o[c] = (_Float16)0.f;
}

// ---------------------------------------------------------------- small helpers
__global__ void cvt_weight_kernel(const float* __restrict__ w, _Float16* __restrict__ o,
                                  int C, int Kp) {
  const int r = blockIdx.x;
  for (int c = threadIdx.x; c < Kp; c += blockDim.x)
    o[(size_t)r * Kp + c] = (c < C) ? (_Float16)w[(size_t)r * C + c] : (_Float16)0.f;
}

__global__ void fill_zero_kernel(float* __restrict__ p, int n) {
  const int i = blockIdx.x * 256 + threadIdx.x;
  if (i < n) p[i] = 0.f;
}

__global__ void cvt_f2h_kernel(const float* __restrict__ f, _Float16* __restrict__ h, int n) {
  const int i = blockIdx.x * 256 + threadIdx.x;
  if (i < n) h[i] = (_Float16)f[i];
}

__global__ void finalize_kernel(const float* __restrict__ partials, int N, float invM,
                                const float* __restrict__ g, const float* __restrict__ b,
                                float* __restrict__ sc, float* __restrict__ sh) {
  const int c = threadIdx.x;
  if (c >= N) return;
  float s = 0.f, s2 = 0.f;
  for (int p = 0; p < 256; ++p) {
    s += partials[2 * ((size_t)p * N + c)];
    s2 += partials[2 * ((size_t)p * N + c) + 1];
  }
  const float mean = s * invM;
  const float var = s2 * invM - mean * mean;
  const float scale = g[c] * rsqrtf(var + 1e-5f);
  sc[c] = scale;
  sh[c] = b[c] - mean * scale;
}

// ---------------------------------------------------------------- async global->LDS staging
// dynamic-LDS offsets start at 0, matching the raw LDS address the async op expects.
__device__ __forceinline__ void async_g2l_b128(unsigned lds_byte_off,
                                               const _Float16* base,
                                               unsigned glb_byte_off) {
  asm volatile("global_load_async_to_lds_b128 %0, %1, %2"
               :: "v"(lds_byte_off), "v"(glb_byte_off), "s"(base)
               : "memory");
}
__device__ __forceinline__ void wait_async0() {
  asm volatile("s_wait_asynccnt 0x0" ::: "memory");
}

// ---------------------------------------------------------------- WMMA GEMM + BN epilogues
// Block: 256 thr = 8 waves; block tile 32(M) x 128(N); each wave: 2 M-subtiles x 16 cols.
// mode 0: stats (sum, sumsq per channel -> partials)
// mode 1: apply BN+ReLU, store f16
// mode 2: apply BN+ReLU, max over kGroup rows, ordered-uint atomicMax to f32
// mode 3: apply BN+ReLU, store f32 (final output)
template <int KT>
__global__ void __launch_bounds__(256) gemm_bn_kernel(
    const _Float16* __restrict__ A, int lda,
    const _Float16* __restrict__ Wt, int ldw,
    int M, int N, int mode,
    float* __restrict__ partials,
    const float* __restrict__ sc, const float* __restrict__ sh,
    _Float16* __restrict__ outH, int ldoH,
    float* __restrict__ outMax, int kGroup, int colOff, int ldoMax,
    float* __restrict__ outF, int ldoF) {
  extern __shared__ _Float16 smem[];
  _Float16* Ah = smem;                          // [32][LDS_STRIDE]
  _Float16* Bh = smem + 32 * LDS_STRIDE;        // [128][LDS_STRIDE]
  const unsigned BH_OFF = 32 * LDS_STRIDE * 2;  // byte offset of Bh

  const int tid = threadIdx.x;
  const int m0 = blockIdx.x * 32;
  const int n0 = blockIdx.y * 128;
  const int lane = tid & 31;
  const int wv = tid >> 5;
  const int mrow = lane & 15;
  const int h = lane >> 4;
  v8f acc0 = {0.f, 0.f, 0.f, 0.f, 0.f, 0.f, 0.f, 0.f};
  v8f acc1 = {0.f, 0.f, 0.f, 0.f, 0.f, 0.f, 0.f, 0.f};

  for (int kc = 0; kc < KT; kc += 128) {
    constexpr int KREM = KT % 128 ? KT % 128 : 128;
    const int kcur = (kc + 128 <= KT) ? 128 : KREM;
    const int nch = kcur >> 3;                  // 16B chunks per row
    // async-stage A tile: 32 rows x kcur
    for (int c = tid; c < 32 * nch; c += 256) {
      const int r = c / nch, cc = c - r * nch;
      async_g2l_b128((unsigned)((r * LDS_STRIDE + cc * 8) * 2), A,
                     (unsigned)((((size_t)(m0 + r) * lda) + kc + cc * 8) * 2));
    }
    // async-stage B tile (weights, Bt row-major): 128 rows x kcur
    for (int c = tid; c < 128 * nch; c += 256) {
      const int r = c / nch, cc = c - r * nch;
      async_g2l_b128(BH_OFF + (unsigned)((r * LDS_STRIDE + cc * 8) * 2), Wt,
                     (unsigned)((((size_t)(n0 + r) * ldw) + kc + cc * 8) * 2));
    }
    wait_async0();
    __syncthreads();
    for (int k0 = 0; k0 < kcur; k0 += 32) {
      Frag fa0, fa1, fb;
      // B 32x16 f16 layout: lane holds Bt[n][k0+h*8 .. +7] and Bt[n][k0+16+h*8 .. +7]
      fb.q[0] = *(const U128*)&Bh[(wv * 16 + mrow) * LDS_STRIDE + k0 + h * 8];
      fb.q[1] = *(const U128*)&Bh[(wv * 16 + mrow) * LDS_STRIDE + k0 + 16 + h * 8];
      // A 16x32 f16 layout: lane holds A[m][k0+h*8 .. +7] and A[m][k0+16+h*8 .. +7]
      fa0.q[0] = *(const U128*)&Ah[mrow * LDS_STRIDE + k0 + h * 8];
      fa0.q[1] = *(const U128*)&Ah[mrow * LDS_STRIDE + k0 + 16 + h * 8];
      fa1.q[0] = *(const U128*)&Ah[(16 + mrow) * LDS_STRIDE + k0 + h * 8];
      fa1.q[1] = *(const U128*)&Ah[(16 + mrow) * LDS_STRIDE + k0 + 16 + h * 8];
      acc0 = __builtin_amdgcn_wmma_f32_16x16x32_f16(false, fa0.v, false, fb.v,
                                                    (short)0, acc0, false, false);
      acc1 = __builtin_amdgcn_wmma_f32_16x16x32_f16(false, fa1.v, false, fb.v,
                                                    (short)0, acc1, false, false);
    }
    __syncthreads();
  }

  // C layout: lane l, element j -> C[(l>>4)*8 + j][l&15]
  const int col = n0 + wv * 16 + mrow;

  if (mode == 0) {
    float s = 0.f, s2 = 0.f;
#pragma unroll
    for (int j = 0; j < 8; ++j) { const float v = acc0[j]; s += v; s2 += v * v; }
#pragma unroll
    for (int j = 0; j < 8; ++j) { const float v = acc1[j]; s += v; s2 += v * v; }
    s += __shfl_xor(s, 16, 32);
    s2 += __shfl_xor(s2, 16, 32);
    if (h == 0) {
      const int part = blockIdx.x & 255;
      float* p = partials + 2 * ((size_t)part * N + col);
      atomicAdd(p, s);
      atomicAdd(p + 1, s2);
    }
  } else if (mode == 1) {
    const float a_ = sc[col], b_ = sh[col];
#pragma unroll
    for (int j = 0; j < 8; ++j) {
      const int r0 = m0 + h * 8 + j;
      outH[(size_t)r0 * ldoH + col] = (_Float16)fmaxf(acc0[j] * a_ + b_, 0.f);
      outH[(size_t)(r0 + 16) * ldoH + col] = (_Float16)fmaxf(acc1[j] * a_ + b_, 0.f);
    }
  } else if (mode == 2) {
    const float a_ = sc[col], b_ = sh[col];
    float mx0 = 0.f, mx1 = 0.f;                 // ReLU output >= 0
#pragma unroll
    for (int j = 0; j < 8; ++j) {
      mx0 = fmaxf(mx0, fmaxf(acc0[j] * a_ + b_, 0.f));
      mx1 = fmaxf(mx1, fmaxf(acc1[j] * a_ + b_, 0.f));
    }
    mx0 = fmaxf(mx0, __shfl_xor(mx0, 16, 32));
    mx1 = fmaxf(mx1, __shfl_xor(mx1, 16, 32));
    if (h == 0) {
      const int s0 = m0 / kGroup;               // 16-row subtile never straddles a group
      const int s1 = (m0 + 16) / kGroup;
      atomicMax((unsigned int*)&outMax[(size_t)s0 * ldoMax + colOff + col],
                __float_as_uint(mx0));
      atomicMax((unsigned int*)&outMax[(size_t)s1 * ldoMax + colOff + col],
                __float_as_uint(mx1));
    }
  } else {
    const float a_ = sc[col], b_ = sh[col];
#pragma unroll
    for (int j = 0; j < 8; ++j) {
      const int r0 = m0 + h * 8 + j;
      outF[(size_t)r0 * ldoF + col] = fmaxf(acc0[j] * a_ + b_, 0.f);
      outF[(size_t)(r0 + 16) * ldoF + col] = fmaxf(acc1[j] * a_ + b_, 0.f);
    }
  }
}

// ---------------------------------------------------------------- launcher
extern "C" void kernel_launch(void* const* d_in, const int* in_sizes, int n_in,
                              void* d_out, int out_size, void* d_ws, size_t ws_size,
                              hipStream_t stream) {
  (void)in_sizes; (void)n_in; (void)out_size; (void)ws_size;
  const float* xyz  = (const float*)d_in[0];
  const float* feat = (const float*)d_in[1];
  const float* w1[2] = {(const float*)d_in[2], (const float*)d_in[8]};
  const float* g1[2] = {(const float*)d_in[3], (const float*)d_in[9]};
  const float* b1[2] = {(const float*)d_in[4], (const float*)d_in[10]};
  const float* w2[2] = {(const float*)d_in[5], (const float*)d_in[11]};
  const float* g2[2] = {(const float*)d_in[6], (const float*)d_in[12]};
  const float* b2[2] = {(const float*)d_in[7], (const float*)d_in[13]};
  const float* wf1 = (const float*)d_in[14];
  const float* gf1 = (const float*)d_in[15];
  const float* bf1 = (const float*)d_in[16];
  const float* wf2 = (const float*)d_in[17];
  const float* gf2 = (const float*)d_in[18];
  const float* bf2 = (const float*)d_in[19];

  float* newxyz = (float*)d_out;                       // (B,1024,3)
  float* outx   = (float*)d_out + (size_t)B_ * S_ * 3; // (B,1024,256)

  char* ws = (char*)d_ws;
  size_t off = 0;
  auto alloc = [&](size_t bytes) -> char* {
    char* p = ws + off;
    off += (bytes + 255) & ~(size_t)255;
    return p;
  };
  const int MMAX = B_ * S_ * 32;
  _Float16* Xh  = (_Float16*)alloc((size_t)MMAX * KP1 * 2);
  _Float16* A1h = (_Float16*)alloc((size_t)MMAX * MID * 2);
  _Float16* Ch  = (_Float16*)alloc((size_t)B_ * S_ * CF * 2);
  float*    Cf  = (float*)   alloc((size_t)B_ * S_ * CF * 4);
  _Float16* F1h = (_Float16*)alloc((size_t)B_ * S_ * OUTC * 2);
  _Float16* w1h[2] = {(_Float16*)alloc(MID * KP1 * 2), (_Float16*)alloc(MID * KP1 * 2)};
  _Float16* w2h[2] = {(_Float16*)alloc(OUTC * MID * 2), (_Float16*)alloc(OUTC * MID * 2)};
  _Float16* wf1h = (_Float16*)alloc(OUTC * CF * 2);
  _Float16* wf2h = (_Float16*)alloc(OUTC * OUTC * 2);
  float* partials = (float*)alloc(256 * 256 * 2 * 4);
  float* scb = (float*)alloc(256 * 4);
  float* shb = (float*)alloc(256 * 4);
  int* idxb  = (int*)alloc((size_t)MMAX * 4);

  fps_kernel<<<dim3(B_), dim3(256), 0, stream>>>(xyz, newxyz);

  cvt_weight_kernel<<<dim3(MID),  dim3(128), 0, stream>>>(w1[0], w1h[0], 67, KP1);
  cvt_weight_kernel<<<dim3(MID),  dim3(128), 0, stream>>>(w1[1], w1h[1], 67, KP1);
  cvt_weight_kernel<<<dim3(OUTC), dim3(128), 0, stream>>>(w2[0], w2h[0], MID, MID);
  cvt_weight_kernel<<<dim3(OUTC), dim3(128), 0, stream>>>(w2[1], w2h[1], MID, MID);
  cvt_weight_kernel<<<dim3(OUTC), dim3(128), 0, stream>>>(wf1, wf1h, CF, CF);
  cvt_weight_kernel<<<dim3(OUTC), dim3(128), 0, stream>>>(wf2, wf2h, OUTC, OUTC);

  fill_zero_kernel<<<dim3((B_ * S_ * CF + 255) / 256), dim3(256), 0, stream>>>(Cf, B_ * S_ * CF);

  const float r2v[2] = {0.01f, 0.04f};
  const int kv[2] = {16, 32};
  for (int s = 0; s < 2; ++s) {
    const int k = kv[s];
    const int M = B_ * S_ * k;
    ball_query_kernel<<<dim3(B_, S_ / 256), dim3(256), 0, stream>>>(xyz, newxyz, idxb, r2v[s], k);
    build_x_kernel<<<dim3(M / 256), dim3(256), 0, stream>>>(xyz, feat, newxyz, idxb, Xh, k);

    // layer 1: (M,96) x (96,128)
    fill_zero_kernel<<<dim3(512), dim3(256), 0, stream>>>(partials, 256 * 256 * 2);
    gemm_bn_kernel<KP1><<<dim3(M / 32, 1), dim3(256), SMEM_BYTES, stream>>>(
        Xh, KP1, w1h[s], KP1, M, MID, 0,
        partials, nullptr, nullptr, nullptr, 0, nullptr, 0, 0, 0, nullptr, 0);
    finalize_kernel<<<dim3(1), dim3(256), 0, stream>>>(partials, MID, 1.0f / (float)M,
                                                       g1[s], b1[s], scb, shb);
    gemm_bn_kernel<KP1><<<dim3(M / 32, 1), dim3(256), SMEM_BYTES, stream>>>(
        Xh, KP1, w1h[s], KP1, M, MID, 1,
        nullptr, scb, shb, A1h, MID, nullptr, 0, 0, 0, nullptr, 0);

    // layer 2: (M,128) x (128,256) + max over k
    fill_zero_kernel<<<dim3(512), dim3(256), 0, stream>>>(partials, 256 * 256 * 2);
    gemm_bn_kernel<MID><<<dim3(M / 32, 2), dim3(256), SMEM_BYTES, stream>>>(
        A1h, MID, w2h[s], MID, M, OUTC, 0,
        partials, nullptr, nullptr, nullptr, 0, nullptr, 0, 0, 0, nullptr, 0);
    finalize_kernel<<<dim3(1), dim3(256), 0, stream>>>(partials, OUTC, 1.0f / (float)M,
                                                       g2[s], b2[s], scb, shb);
    gemm_bn_kernel<MID><<<dim3(M / 32, 2), dim3(256), SMEM_BYTES, stream>>>(
        A1h, MID, w2h[s], MID, M, OUTC, 2,
        nullptr, scb, shb, nullptr, 0, Cf, k, s * OUTC, CF, nullptr, 0);
  }

  cvt_f2h_kernel<<<dim3((B_ * S_ * CF + 255) / 256), dim3(256), 0, stream>>>(Cf, Ch, B_ * S_ * CF);

  const int Mf = B_ * S_;
  // fuse 1: (8192,512) x (512,256)
  fill_zero_kernel<<<dim3(512), dim3(256), 0, stream>>>(partials, 256 * 256 * 2);
  gemm_bn_kernel<CF><<<dim3(Mf / 32, 2), dim3(256), SMEM_BYTES, stream>>>(
      Ch, CF, wf1h, CF, Mf, OUTC, 0,
      partials, nullptr, nullptr, nullptr, 0, nullptr, 0, 0, 0, nullptr, 0);
  finalize_kernel<<<dim3(1), dim3(256), 0, stream>>>(partials, OUTC, 1.0f / (float)Mf,
                                                     gf1, bf1, scb, shb);
  gemm_bn_kernel<CF><<<dim3(Mf / 32, 2), dim3(256), SMEM_BYTES, stream>>>(
      Ch, CF, wf1h, CF, Mf, OUTC, 1,
      nullptr, scb, shb, F1h, OUTC, nullptr, 0, 0, 0, nullptr, 0);

  // fuse 2: (8192,256) x (256,256) -> final output
  fill_zero_kernel<<<dim3(512), dim3(256), 0, stream>>>(partials, 256 * 256 * 2);
  gemm_bn_kernel<OUTC><<<dim3(Mf / 32, 2), dim3(256), SMEM_BYTES, stream>>>(
      F1h, OUTC, wf2h, OUTC, Mf, OUTC, 0,
      partials, nullptr, nullptr, nullptr, 0, nullptr, 0, 0, 0, nullptr, 0);
  finalize_kernel<<<dim3(1), dim3(256), 0, stream>>>(partials, OUTC, 1.0f / (float)Mf,
                                                     gf2, bf2, scb, shb);
  gemm_bn_kernel<OUTC><<<dim3(Mf / 32, 2), dim3(256), SMEM_BYTES, stream>>>(
      F1h, OUTC, wf2h, OUTC, Mf, OUTC, 3,
      nullptr, scb, shb, nullptr, 0, nullptr, 0, 0, 0, outx, OUTC);
}
